// MatMulModel_70317204570841
// MI455X (gfx1250) — matrix operand order repaired and checked by reading, not compiled, not run on hardware
//
#include <hip/hip_runtime.h>

// ---------------------------------------------------------------------------
// out[M,N] = W[M,K] @ X[K,N] + bias[N]   (fp32 in/out), M=N=2000, K=8192.
//
// Strategy (MI455X / gfx1250):
//  * Pre-pass (once per launch, into d_ws): split fp32 -> (hi,lo) bf16 pair
//    (bf16x3 emulation, ~2^-16 rel err) and transpose X so both operands are
//    row-linear bf16 [row, K]. Padded to 2048 rows -> no load guards in GEMM.
//  * Packed GEMM: 128x128x32 tiles, 8 wave32 waves, double-buffered LDS
//    staging via GLOBAL_LOAD_ASYNC_TO_LDS_B128 (ASYNCcnt), fragments via
//    ds_load_b128, 24 v_wmma_f32_16x16x32_bf16 per K-step.
//  * Fallback fused kernel (conversion in-loop) if ws_size is too small.
// ---------------------------------------------------------------------------

typedef __bf16          v16bf __attribute__((ext_vector_type(16)));
typedef float           v8f   __attribute__((ext_vector_type(8)));
typedef float           f32x4 __attribute__((ext_vector_type(4)));
typedef unsigned int    u32x4 __attribute__((ext_vector_type(4)));
typedef unsigned short  u16x4 __attribute__((ext_vector_type(4)));
typedef int             v4i   __attribute__((ext_vector_type(4)));

#define M_DIM 2000
#define N_DIM 2000
#define K_DIM 8192
#define MP    2048        // padded rows for W and X^T
#define BM    128
#define BN    128
#define BK    32
#define LDT   (BK + 8)    // LDS row stride (elems): 40 -> 80 B (16B aligned)
#define TSZ   (BM * LDT)  // elems per LDS tile array (5120)
#define SMEM_BYTES (2 /*bufs*/ * 4 /*arrays*/ * TSZ * 2 /*bytes*/)  // 81920

__device__ __forceinline__ unsigned short f32_to_bf16_rne(float f) {
    unsigned int u = __float_as_uint(f);
    u += 0x7FFFu + ((u >> 16) & 1u);
    return (unsigned short)(u >> 16);
}
__device__ __forceinline__ float bf16_to_f32(unsigned short h) {
    return __uint_as_float(((unsigned int)h) << 16);
}

union FragBF {
    struct { u32x4 a, b; } q;
    v16bf v;
};

// ---- CDNA5 async global->LDS (ASYNCcnt) with compile-safe fallback --------
// Probe-learned signature: (int4 AS1* gsrc, int4 AS3* ldst, Ii offset, Ii cpol)
#if __has_builtin(__builtin_amdgcn_global_load_async_to_lds_b128)
#define ASYNC_LDS_OK 1
typedef __attribute__((address_space(1))) v4i* as1_v4i;
typedef __attribute__((address_space(3))) v4i* as3_v4i;
#endif

__device__ __forceinline__ void async_copy16(const unsigned short* g, unsigned short* l) {
#ifdef ASYNC_LDS_OK
    __builtin_amdgcn_global_load_async_to_lds_b128(
        (as1_v4i)(v4i*)g, (as3_v4i)(v4i*)l, 0, 0);
#else
    *(u32x4*)l = *(const u32x4*)g;   // sync fallback: global load + ds store
#endif
}

__device__ __forceinline__ void wait_async_done() {
#ifdef ASYNC_LDS_OK
#if __has_builtin(__builtin_amdgcn_s_wait_asynccnt)
    __builtin_amdgcn_s_wait_asynccnt(0);
#else
    asm volatile("s_wait_asynccnt 0x0" ::: "memory");
#endif
#endif
}

// ===========================================================================
// Pre-pass 1: W[M,K] f32 -> Whi/Wlo[MP,K] bf16 (rows >= M zero-filled)
// ===========================================================================
__global__ __launch_bounds__(256)
void split_w(const float* __restrict__ W,
             unsigned short* __restrict__ Whi,
             unsigned short* __restrict__ Wlo)
{
    size_t e   = ((size_t)blockIdx.x * 256 + threadIdx.x) * 4;  // elem index
    int    row = (int)(e >> 13);          // / 8192
    int    col = (int)(e & (K_DIM - 1));
    u16x4 hi = {0, 0, 0, 0}, lo = {0, 0, 0, 0};
    if (row < M_DIM) {
        f32x4 v = *(const f32x4*)(W + (size_t)row * K_DIM + col);
#pragma unroll
        for (int j = 0; j < 4; ++j) {
            unsigned short h = f32_to_bf16_rne(v[j]);
            hi[j] = h;
            lo[j] = f32_to_bf16_rne(v[j] - bf16_to_f32(h));
        }
    }
    *(u16x4*)(Whi + (size_t)row * K_DIM + col) = hi;
    *(u16x4*)(Wlo + (size_t)row * K_DIM + col) = lo;
}

// ===========================================================================
// Pre-pass 2: X[K,N] f32 -> XThi/XTlo[NP=MP,K] bf16 transposed, zero-padded
// ===========================================================================
__global__ __launch_bounds__(256)
void split_transpose_x(const float* __restrict__ X,
                       unsigned short* __restrict__ XThi,
                       unsigned short* __restrict__ XTlo)
{
    __shared__ unsigned short thi[32][34];   // +2 pad: kill bank conflicts
    __shared__ unsigned short tlo[32][34];
    const int k0 = blockIdx.x * 32;          // K/32 = 256 blocks
    const int n0 = blockIdx.y * 32;          // MP/32 = 64 blocks
    const int tx = threadIdx.x;              // 0..31
    const int ty = threadIdx.y;              // 0..7
#pragma unroll
    for (int j = 0; j < 4; ++j) {
        int k = ty + j * 8;
        int n = n0 + tx;
        float v = (n < N_DIM) ? X[(size_t)(k0 + k) * N_DIM + n] : 0.f;
        unsigned short h = f32_to_bf16_rne(v);
        thi[k][tx] = h;
        tlo[k][tx] = f32_to_bf16_rne(v - bf16_to_f32(h));
    }
    __syncthreads();
#pragma unroll
    for (int j = 0; j < 4; ++j) {
        int n = ty + j * 8;                  // row of X^T within tile
        XThi[(size_t)(n0 + n) * K_DIM + k0 + tx] = thi[tx][n];
        XTlo[(size_t)(n0 + n) * K_DIM + k0 + tx] = tlo[tx][n];
    }
}

// ===========================================================================
// Packed GEMM: bf16x3 on v_wmma_f32_16x16x32_bf16, async double-buffered LDS
// ===========================================================================
__global__ __launch_bounds__(256)
void gemm_packed_wmma(const unsigned short* __restrict__ Whi,
                      const unsigned short* __restrict__ Wlo,
                      const unsigned short* __restrict__ XThi,
                      const unsigned short* __restrict__ XTlo,
                      const float* __restrict__ bias,
                      float* __restrict__ Out)
{
    extern __shared__ unsigned short dsm[];  // 2 bufs x {Ahi,Alo,Bhi,Blo} x TSZ

    const int tid  = threadIdx.x;
    const int lane = tid & 31;
    const int wave = tid >> 5;
    const int wm   = wave & 3;               // 4 waves along M (32 rows each)
    const int wn   = wave >> 2;              // 2 waves along N (64 cols each)
    const int half = lane >> 4;
    const int lm   = lane & 15;

    const int m0 = blockIdx.y * BM;
    const int n0 = blockIdx.x * BN;

    // staging slice: each thread copies 16 elems (32 B) of one 32-elem row
    const int srow = tid >> 1;               // 0..127
    const int se0  = (tid & 1) * 16;         // elem offset in row

    v8f acc[2][4];
#pragma unroll
    for (int t = 0; t < 2; ++t)
#pragma unroll
        for (int nt = 0; nt < 4; ++nt)
            acc[t][nt] = (v8f){0.f, 0.f, 0.f, 0.f, 0.f, 0.f, 0.f, 0.f};

    auto stage = [&](int b, int kk) {
        unsigned short* aH = dsm + (b * 4 + 0) * TSZ;
        unsigned short* aL = dsm + (b * 4 + 1) * TSZ;
        unsigned short* bH = dsm + (b * 4 + 2) * TSZ;
        unsigned short* bL = dsm + (b * 4 + 3) * TSZ;
        const size_t gA = (size_t)(m0 + srow) * K_DIM + kk + se0;  // rows < MP
        const size_t gB = (size_t)(n0 + srow) * K_DIM + kk + se0;
        const int    ll = srow * LDT + se0;
        async_copy16(Whi  + gA,     aH + ll);
        async_copy16(Whi  + gA + 8, aH + ll + 8);
        async_copy16(Wlo  + gA,     aL + ll);
        async_copy16(Wlo  + gA + 8, aL + ll + 8);
        async_copy16(XThi + gB,     bH + ll);
        async_copy16(XThi + gB + 8, bH + ll + 8);
        async_copy16(XTlo + gB,     bL + ll);
        async_copy16(XTlo + gB + 8, bL + ll + 8);
    };

    const int KT = K_DIM / BK;               // 256
    stage(0, 0);                             // prologue

    for (int kt = 0; kt < KT; ++kt) {
        const int b = kt & 1;
        wait_async_done();                   // buffer b's DMA complete (this wave)
        __syncthreads();                     // all waves' slices visible
        if (kt + 1 < KT)
            stage(1 - b, (kt + 1) * BK);     // overlap next tile DMA w/ compute

        const unsigned short* aH  = dsm + (b * 4 + 0) * TSZ;
        const unsigned short* aL  = dsm + (b * 4 + 1) * TSZ;
        const unsigned short* bHs = dsm + (b * 4 + 2) * TSZ;
        const unsigned short* bLs = dsm + (b * 4 + 3) * TSZ;

        // A 16x32 bf16 frag: lanes 0-15 K{0-7,16-23}; lanes 16-31 K{8-15,24-31}
        FragBF aHi[2], aLo[2];
#pragma unroll
        for (int t = 0; t < 2; ++t) {
            const unsigned short* ph = aH + (wm * 32 + t * 16 + lm) * LDT;
            const unsigned short* pl = aL + (wm * 32 + t * 16 + lm) * LDT;
            aHi[t].q.a = *(const u32x4*)(ph + half * 8);
            aHi[t].q.b = *(const u32x4*)(ph + 16 + half * 8);
            aLo[t].q.a = *(const u32x4*)(pl + half * 8);
            aLo[t].q.b = *(const u32x4*)(pl + 16 + half * 8);
        }
        // B 32x16 bf16 frag: lanes 0-15 K{0-7,16-23}; lanes 16-31 K{8-15,24-31}
        FragBF bHi[4], bLo[4];
#pragma unroll
        for (int nt = 0; nt < 4; ++nt) {
            const unsigned short* ph = bHs + (wn * 64 + nt * 16 + lm) * LDT;
            const unsigned short* pl = bLs + (wn * 64 + nt * 16 + lm) * LDT;
            bHi[nt].q.a = *(const u32x4*)(ph + half * 8);
            bHi[nt].q.b = *(const u32x4*)(ph + 16 + half * 8);
            bLo[nt].q.a = *(const u32x4*)(pl + half * 8);
            bLo[nt].q.b = *(const u32x4*)(pl + 16 + half * 8);
        }

#pragma unroll
        for (int t = 0; t < 2; ++t) {
#pragma unroll
            for (int nt = 0; nt < 4; ++nt) {
                v8f c = acc[t][nt];
                c = __builtin_amdgcn_wmma_f32_16x16x32_bf16(
                        false, aHi[t].v, false, bHi[nt].v, (short)0, c, false, false);
                c = __builtin_amdgcn_wmma_f32_16x16x32_bf16(
                        false, aHi[t].v, false, bLo[nt].v, (short)0, c, false, false);
                c = __builtin_amdgcn_wmma_f32_16x16x32_bf16(
                        false, aLo[t].v, false, bHi[nt].v, (short)0, c, false, false);
                acc[t][nt] = c;
            }
        }
    }

    // D lane map: VGPR r -> M = r + 8*half, N = lane&15; bias over last dim
#pragma unroll
    for (int t = 0; t < 2; ++t) {
#pragma unroll
        for (int nt = 0; nt < 4; ++nt) {
            int gn = n0 + wn * 64 + nt * 16 + lm;
            float bv = (gn < N_DIM) ? bias[gn] : 0.f;
#pragma unroll
            for (int r = 0; r < 8; ++r) {
                int gm = m0 + wm * 32 + t * 16 + half * 8 + r;
                if (gm < M_DIM && gn < N_DIM)
                    Out[(size_t)gm * N_DIM + gn] = acc[t][nt][r] + bv;
            }
        }
    }
}

// ===========================================================================
// Fallback: fused conversion GEMM (used when ws_size is too small)
// ===========================================================================
__global__ __launch_bounds__(256)
void gemm_bf16x3_fused(const float* __restrict__ W,
                       const float* __restrict__ bias,
                       const float* __restrict__ X,
                       float* __restrict__ Out)
{
    __shared__ __align__(16) unsigned short sAhi[BM * LDT];
    __shared__ __align__(16) unsigned short sAlo[BM * LDT];
    __shared__ __align__(16) unsigned short sBhi[BN * LDT];
    __shared__ __align__(16) unsigned short sBlo[BN * LDT];

    const int tid  = threadIdx.x;
    const int lane = tid & 31;
    const int wave = tid >> 5;
    const int wm   = wave & 3;
    const int wn   = wave >> 2;
    const int half = lane >> 4;
    const int lm   = lane & 15;
    const int m0 = blockIdx.y * BM;
    const int n0 = blockIdx.x * BN;

    v8f acc[2][4];
#pragma unroll
    for (int t = 0; t < 2; ++t)
#pragma unroll
        for (int nt = 0; nt < 4; ++nt)
            acc[t][nt] = (v8f){0.f, 0.f, 0.f, 0.f, 0.f, 0.f, 0.f, 0.f};

    for (int k0 = 0; k0 < K_DIM; k0 += BK) {
        if (k0 + BK < K_DIM) {
            int prow = m0 + (tid >> 3);
            if (prow >= M_DIM) prow = M_DIM - 1;
            __builtin_prefetch(W + (size_t)prow * K_DIM + (k0 + BK) + ((tid & 7) << 2), 0, 3);
        }
#pragma unroll
        for (int i = 0; i < 4; ++i) {
            int idx = tid + i * 256;
            int row = idx >> 3;
            int kq  = (idx & 7) << 2;
            int gr  = m0 + row;
            f32x4 v = {0.f, 0.f, 0.f, 0.f};
            if (gr < M_DIM)
                v = *(const f32x4*)(W + (size_t)gr * K_DIM + k0 + kq);
            u16x4 hi, lo;
#pragma unroll
            for (int j = 0; j < 4; ++j) {
                unsigned short h = f32_to_bf16_rne(v[j]);
                hi[j] = h;
                lo[j] = f32_to_bf16_rne(v[j] - bf16_to_f32(h));
            }
            *(u16x4*)&sAhi[row * LDT + kq] = hi;
            *(u16x4*)&sAlo[row * LDT + kq] = lo;
        }
#pragma unroll
        for (int i = 0; i < 4; ++i) {
            int idx = tid + i * 256;
            int k   = idx >> 5;
            int nq  = (idx & 31) << 2;
            int gn  = n0 + nq;
            const float* src = X + (size_t)(k0 + k) * N_DIM + gn;
            f32x4 v = {0.f, 0.f, 0.f, 0.f};
            if (gn + 3 < N_DIM) {
                v = *(const f32x4*)src;
            } else {
                if (gn + 0 < N_DIM) v[0] = src[0];
                if (gn + 1 < N_DIM) v[1] = src[1];
                if (gn + 2 < N_DIM) v[2] = src[2];
            }
#pragma unroll
            for (int j = 0; j < 4; ++j) {
                unsigned short h = f32_to_bf16_rne(v[j]);
                sBhi[(nq + j) * LDT + k] = h;
                sBlo[(nq + j) * LDT + k] = f32_to_bf16_rne(v[j] - bf16_to_f32(h));
            }
        }
        __syncthreads();

        FragBF aHi[2], aLo[2], bHi[4], bLo[4];
#pragma unroll
        for (int t = 0; t < 2; ++t) {
            const unsigned short* ph = &sAhi[(wm * 32 + t * 16 + lm) * LDT];
            const unsigned short* pl = &sAlo[(wm * 32 + t * 16 + lm) * LDT];
            aHi[t].q.a = *(const u32x4*)(ph + half * 8);
            aHi[t].q.b = *(const u32x4*)(ph + 16 + half * 8);
            aLo[t].q.a = *(const u32x4*)(pl + half * 8);
            aLo[t].q.b = *(const u32x4*)(pl + 16 + half * 8);
        }
#pragma unroll
        for (int nt = 0; nt < 4; ++nt) {
            const unsigned short* ph = &sBhi[(wn * 64 + nt * 16 + lm) * LDT];
            const unsigned short* pl = &sBlo[(wn * 64 + nt * 16 + lm) * LDT];
            bHi[nt].q.a = *(const u32x4*)(ph + half * 8);
            bHi[nt].q.b = *(const u32x4*)(ph + 16 + half * 8);
            bLo[nt].q.a = *(const u32x4*)(pl + half * 8);
            bLo[nt].q.b = *(const u32x4*)(pl + 16 + half * 8);
        }
#pragma unroll
        for (int t = 0; t < 2; ++t) {
#pragma unroll
            for (int nt = 0; nt < 4; ++nt) {
                v8f c = acc[t][nt];
                c = __builtin_amdgcn_wmma_f32_16x16x32_bf16(
                        false, aHi[t].v, false, bHi[nt].v, (short)0, c, false, false);
                c = __builtin_amdgcn_wmma_f32_16x16x32_bf16(
                        false, aHi[t].v, false, bLo[nt].v, (short)0, c, false, false);
                c = __builtin_amdgcn_wmma_f32_16x16x32_bf16(
                        false, aLo[t].v, false, bHi[nt].v, (short)0, c, false, false);
                acc[t][nt] = c;
            }
        }
        __syncthreads();
    }

#pragma unroll
    for (int t = 0; t < 2; ++t) {
#pragma unroll
        for (int nt = 0; nt < 4; ++nt) {
            int gn = n0 + wn * 64 + nt * 16 + lm;
            float bv = (gn < N_DIM) ? bias[gn] : 0.f;
#pragma unroll
            for (int r = 0; r < 8; ++r) {
                int gm = m0 + wm * 32 + t * 16 + half * 8 + r;
                if (gm < M_DIM && gn < N_DIM)
                    Out[(size_t)gm * N_DIM + gn] = acc[t][nt][r] + bv;
            }
        }
    }
}

// ===========================================================================
extern "C" void kernel_launch(void* const* d_in, const int* in_sizes, int n_in,
                              void* d_out, int out_size, void* d_ws, size_t ws_size,
                              hipStream_t stream) {
    const float* W    = (const float*)d_in[0];   // [2000, 8192]
    const float* bias = (const float*)d_in[1];   // [2000]
    const float* X    = (const float*)d_in[2];   // [8192, 2000]
    float*       Out  = (float*)d_out;           // [2000, 2000]

    const size_t halfElems = (size_t)MP * K_DIM;           // 16.78M
    const size_t need      = 4 * halfElems * sizeof(unsigned short);  // 134 MB

    dim3 grid((N_DIM + BN - 1) / BN, (M_DIM + BM - 1) / BM);  // 16 x 16
    dim3 block(256);

    if (ws_size >= need) {
        unsigned short* Whi  = (unsigned short*)d_ws;
        unsigned short* Wlo  = Whi + halfElems;
        unsigned short* XThi = Wlo + halfElems;
        unsigned short* XTlo = XThi + halfElems;

        split_w<<<(unsigned)(halfElems / 4 / 256), 256, 0, stream>>>(W, Whi, Wlo);
        split_transpose_x<<<dim3(K_DIM / 32, MP / 32), dim3(32, 8), 0, stream>>>(X, XThi, XTlo);
        gemm_packed_wmma<<<grid, block, SMEM_BYTES, stream>>>(Whi, Wlo, XThi, XTlo, bias, Out);
    } else {
        gemm_bf16x3_fused<<<grid, block, 0, stream>>>(W, bias, X, Out);
    }
}
